// SPNN_25984552140952
// MI455X (gfx1250) — matrix operand order repaired and checked by reading, not compiled, not run on hardware
//
#include <hip/hip_runtime.h>
#include <hip/hip_bf16.h>

// ---------------- problem constants (fixed by the reference) ----------------
#define HID   256          // hidden channels H
#define GEO   13           // geo feature width
#define DIN   781          // 3*H + G
#define KPAD0 800          // DIN padded up to a multiple of 32 (25 K-steps)
#define DEPTH 3            // hidden layers per branch
#define NBR   4            // branches
#define TILE_M 32          // rows (triplets) per workgroup (2 WMMA M-tiles)

// d_ws layout (elements)
#define W0T_ELEMS (NBR * HID * KPAD0)          // 819200 bf16
#define WHT_ELEMS (NBR * DEPTH * HID * HID)    // 786432 bf16
#define B0F_ELEMS (NBR * HID)                  // 1024 f32
#define BHF_ELEMS (NBR * DEPTH * HID)          // 3072 f32
// then routing metadata: int cnt[4], cursor[4], off[4], tbase[5] (padded to 32)
// then int perm[E]

typedef __attribute__((ext_vector_type(16))) __bf16 v16bf;
typedef __attribute__((ext_vector_type(8)))  __bf16 v8bf;
typedef __attribute__((ext_vector_type(8)))  float  v8f;

__device__ __forceinline__ v16bf frag16(const __bf16* p0, const __bf16* p1) {
  v8bf lo = *(const v8bf*)p0;   // 16B load
  v8bf hi = *(const v8bf*)p1;   // 16B load
  return __builtin_shufflevector(lo, hi, 0,1,2,3,4,5,6,7,8,9,10,11,12,13,14,15);
}

__device__ __forceinline__ int branch_of(int eij, int ejk, int NE) {
  return ((eij >= NE) ? 2 : 0) | ((ejk >= NE) ? 1 : 0);
}

// ---------------------------------------------------------------------------
// Prep: fold BatchNorm (eval) into weights/biases, convert to bf16, transpose
// to [N][K] (K contiguous) so each lane's B-fragment is two 16-byte loads.
// ---------------------------------------------------------------------------
__global__ __launch_bounds__(256) void spnn_prep(
    const float* __restrict__ W0,  const float* __restrict__ b0,
    const float* __restrict__ Wh,  const float* __restrict__ bh,
    const float* __restrict__ gam, const float* __restrict__ bet,
    const float* __restrict__ rmu, const float* __restrict__ rvr,
    __bf16* __restrict__ W0t, __bf16* __restrict__ Wht,
    float* __restrict__ B0f, float* __restrict__ Bhf)
{
  int idx = blockIdx.x * 256 + threadIdx.x;
  if (idx < W0T_ELEMS) {
    int br  = idx / (HID * KPAD0);
    int rem = idx % (HID * KPAD0);
    int n = rem / KPAD0, k = rem % KPAD0;
    int gi = (br * (DEPTH + 1) + 0) * HID + n;                 // layer-0 BN
    float s = gam[gi] * rsqrtf(rvr[gi] + 1e-5f);
    float w = (k < DIN) ? W0[((size_t)(br * DIN + k)) * HID + n] * s : 0.f;
    W0t[((size_t)(br * HID + n)) * KPAD0 + k] = (__bf16)w;
  } else if (idx < W0T_ELEMS + WHT_ELEMS) {
    int r   = idx - W0T_ELEMS;
    int br  = r / (DEPTH * HID * HID);
    int rem = r % (DEPTH * HID * HID);
    int l = rem / (HID * HID);
    int rem2 = rem % (HID * HID);
    int n = rem2 / HID, k = rem2 % HID;
    int gi = (br * (DEPTH + 1) + l + 1) * HID + n;
    float s = gam[gi] * rsqrtf(rvr[gi] + 1e-5f);
    float w = Wh[(((size_t)(br * DEPTH + l)) * HID + k) * HID + n] * s;
    Wht[(((size_t)(br * DEPTH + l)) * HID + n) * HID + k] = (__bf16)w;
  } else if (idx < W0T_ELEMS + WHT_ELEMS + B0F_ELEMS) {
    int r = idx - W0T_ELEMS - WHT_ELEMS;           // br*HID + n
    int br = r / HID, n = r % HID;
    int gi = (br * (DEPTH + 1)) * HID + n;
    float s = gam[gi] * rsqrtf(rvr[gi] + 1e-5f);
    B0f[r] = (b0[r] - rmu[gi]) * s + bet[gi];
  } else if (idx < W0T_ELEMS + WHT_ELEMS + B0F_ELEMS + BHF_ELEMS) {
    int r = idx - W0T_ELEMS - WHT_ELEMS - B0F_ELEMS;   // (br*DEPTH+l)*HID+n
    int br = r / (DEPTH * HID);
    int rem = r % (DEPTH * HID);
    int l = rem / HID, n = rem % HID;
    int gi = (br * (DEPTH + 1) + l + 1) * HID + n;
    float s = gam[gi] * rsqrtf(rvr[gi] + 1e-5f);
    Bhf[r] = (bh[r] - rmu[gi]) * s + bet[gi];
  }
}

// ---------------------------------------------------------------------------
// Routing: bucket rows by branch so each GEMM tile runs exactly one branch.
// meta: [0..3]=cnt  [4..7]=cursor  [8..11]=off  [12..16]=tile base (tb[4]=end)
// ---------------------------------------------------------------------------
__global__ __launch_bounds__(256) void route_count(
    const int* __restrict__ edx_ij, const int* __restrict__ edx_jk,
    const int* __restrict__ num_edge_inside, int* __restrict__ meta, int E)
{
  int idx = blockIdx.x * 256 + threadIdx.x;
  if (idx < E) {
    int NE = num_edge_inside[0];
    atomicAdd(&meta[branch_of(edx_ij[idx], edx_jk[idx], NE)], 1);
  }
}

__global__ void route_scan(int* __restrict__ meta)
{
  if (threadIdx.x == 0 && blockIdx.x == 0) {
    int o = 0, t = 0;
    for (int br = 0; br < NBR; ++br) {
      meta[8 + br]  = o;                         // row offset of bucket
      meta[12 + br] = t;                         // first tile of bucket
      o += meta[br];
      t += (meta[br] + TILE_M - 1) / TILE_M;
    }
    meta[16] = t;                                // total tiles
  }
}

__global__ __launch_bounds__(256) void route_scatter(
    const int* __restrict__ edx_ij, const int* __restrict__ edx_jk,
    const int* __restrict__ num_edge_inside,
    int* __restrict__ meta, int* __restrict__ perm, int E)
{
  int idx = blockIdx.x * 256 + threadIdx.x;
  if (idx < E) {
    int NE = num_edge_inside[0];
    int br = branch_of(edx_ij[idx], edx_jk[idx], NE);
    int pos = atomicAdd(&meta[4 + br], 1);
    perm[meta[8 + br] + pos] = idx;
  }
}

// ---------------------------------------------------------------------------
// Fused gather + single-branch MLP (WMMA bf16) + scatter-add.
// One workgroup = 32 rows (one branch bucket), 8 wave32s.
// Each wave owns a 32x32 output tile: 2 M-tiles x 2 N-tiles, so every
// B fragment feeds two WMMAs (4 WMMAs per K-step per wave).
// ---------------------------------------------------------------------------
__global__ __launch_bounds__(256) void spnn_fused(
    const float* __restrict__ node_feature,
    const float* __restrict__ geo,
    const int*   __restrict__ eidx,        // [3][E]
    const float* __restrict__ att,
    const __bf16* __restrict__ W0t,
    const __bf16* __restrict__ Wht,
    const float* __restrict__ B0f,
    const float* __restrict__ Bhf,
    const int*   __restrict__ meta,
    const int*   __restrict__ perm,
    float* __restrict__ out,
    int E)
{
  __shared__ __bf16 lds_x[TILE_M * KPAD0];      // 50 KB  (x tile, bf16)
  __shared__ __bf16 lds_h[2][TILE_M * HID];     // 32 KB  (ping-pong acts)
  __shared__ int s_i[TILE_M], s_j[TILE_M], s_k[TILE_M];

  // ---- which bucket / rows does this block own? (uniform) ----
  const int b = blockIdx.x;
  if (b >= meta[16]) return;                                   // no work
  const int br = (b >= meta[15]) ? 3 : (b >= meta[14]) ? 2
               : (b >= meta[13]) ? 1 : 0;
  const int rstart = meta[8 + br] + (b - meta[12 + br]) * TILE_M;
  const int rend   = meta[8 + br] + meta[br];                  // off+cnt

  const int tid = threadIdx.x;

  // ---- per-row triplet indices (s_i < 0 marks a dead row) ----
  if (tid < TILE_M) {
    int rg = rstart + tid;
    if (rg < rend) {
      int row = perm[rg];
      s_i[tid] = eidx[row];
      s_j[tid] = eidx[E + row];
      s_k[tid] = eidx[2 * E + row];
    } else {
      s_i[tid] = -1; s_j[tid] = 0; s_k[tid] = 0;
    }
  }
  __syncthreads();

  // ---- stage x = [nf[i] | nf[j] | nf[k] | geo | 0-pad] as bf16 ----
  for (int q = tid; q < TILE_M * 192; q += 256) {   // 3*256 floats as float4
    int r = q / 192, c4 = q % 192;
    float4 v = make_float4(0.f, 0.f, 0.f, 0.f);
    if (s_i[r] >= 0) {
      int sect = c4 >> 6;                 // 0: i, 1: j, 2: k
      int c4l  = c4 & 63;
      int node = (sect == 0) ? s_i[r] : (sect == 1) ? s_j[r] : s_k[r];
      v = ((const float4*)node_feature)[(size_t)node * (HID / 4) + c4l];
    }
    __bf16* dst = &lds_x[r * KPAD0 + c4 * 4];
    dst[0] = (__bf16)v.x; dst[1] = (__bf16)v.y;
    dst[2] = (__bf16)v.z; dst[3] = (__bf16)v.w;
  }
  for (int q = tid; q < TILE_M * 32; q += 256) {    // cols 768..799
    int r = q >> 5, c = (3 * HID) + (q & 31);
    float v = 0.f;
    if (s_i[r] >= 0 && c < DIN) {
      int rg = rstart + r;                // careful: need original row
      int row = perm[rg];
      v = geo[(size_t)row * GEO + (c - 3 * HID)];
    }
    lds_x[r * KPAD0 + c] = (__bf16)v;
  }
  __syncthreads();

  const int lane = tid & 31;
  const int wave = tid >> 5;
  const int half = lane >> 4;     // lane half selects K sub-range
  const int nlo  = lane & 15;     // A row / B,C,D column within 16-tile
  const int hoff = half * 8;      // A-frag K offset & C/D row offset

  const int n0 = (wave * 2)     * 16 + nlo;   // this wave's two N columns
  const int n1 = (wave * 2 + 1) * 16 + nlo;
  const float attb = att[br];

  // -------- layer 0: [32 x 800] @ [800 x 256] --------
  v8f a00 = {0.f,0.f,0.f,0.f,0.f,0.f,0.f,0.f};
  v8f a01 = a00, a10 = a00, a11 = a00;
  {
    const __bf16* Wb0 = W0t + (size_t)br * HID * KPAD0;
    for (int kb = 0; kb < KPAD0 / 32; ++kb) {
      const __bf16* ap0 = &lds_x[nlo * KPAD0 + kb * 32 + hoff];
      const __bf16* ap1 = ap0 + 16 * KPAD0;
      v16bf fA0 = frag16(ap0, ap0 + 16);
      v16bf fA1 = frag16(ap1, ap1 + 16);
      const __bf16* bp0 = Wb0 + (size_t)n0 * KPAD0 + kb * 32 + hoff;
      const __bf16* bp1 = Wb0 + (size_t)n1 * KPAD0 + kb * 32 + hoff;
      v16bf fB0 = frag16(bp0, bp0 + 16);
      v16bf fB1 = frag16(bp1, bp1 + 16);
      a00 = __builtin_amdgcn_wmma_f32_16x16x32_bf16(false, fA0, false, fB0,
                                                    (short)0, a00, false, false);
      a01 = __builtin_amdgcn_wmma_f32_16x16x32_bf16(false, fA0, false, fB1,
                                                    (short)0, a01, false, false);
      a10 = __builtin_amdgcn_wmma_f32_16x16x32_bf16(false, fA1, false, fB0,
                                                    (short)0, a10, false, false);
      a11 = __builtin_amdgcn_wmma_f32_16x16x32_bf16(false, fA1, false, fB1,
                                                    (short)0, a11, false, false);
    }
    float bias0 = B0f[br * HID + n0];
    float bias1 = B0f[br * HID + n1];
    #pragma unroll
    for (int v = 0; v < 8; ++v) {
      int M0 = v + hoff, M1 = 16 + v + hoff;
      lds_h[0][M0 * HID + n0] = (__bf16)fmaxf(a00[v] + bias0, 0.f);
      lds_h[0][M0 * HID + n1] = (__bf16)fmaxf(a01[v] + bias1, 0.f);
      lds_h[0][M1 * HID + n0] = (__bf16)fmaxf(a10[v] + bias0, 0.f);
      lds_h[0][M1 * HID + n1] = (__bf16)fmaxf(a11[v] + bias1, 0.f);
    }
  }
  __syncthreads();

  // -------- hidden layers: [32 x 256] @ [256 x 256] --------
  int bufin = 0;
  for (int l = 0; l < DEPTH; ++l) {
    v8f d00 = {0.f,0.f,0.f,0.f,0.f,0.f,0.f,0.f};
    v8f d01 = d00, d10 = d00, d11 = d00;
    const __bf16* Wb = Wht + (size_t)(br * DEPTH + l) * HID * HID;
    for (int kb = 0; kb < HID / 32; ++kb) {
      const __bf16* ap0 = &lds_h[bufin][nlo * HID + kb * 32 + hoff];
      const __bf16* ap1 = ap0 + 16 * HID;
      v16bf fA0 = frag16(ap0, ap0 + 16);
      v16bf fA1 = frag16(ap1, ap1 + 16);
      const __bf16* bp0 = Wb + (size_t)n0 * HID + kb * 32 + hoff;
      const __bf16* bp1 = Wb + (size_t)n1 * HID + kb * 32 + hoff;
      v16bf fB0 = frag16(bp0, bp0 + 16);
      v16bf fB1 = frag16(bp1, bp1 + 16);
      d00 = __builtin_amdgcn_wmma_f32_16x16x32_bf16(false, fA0, false, fB0,
                                                    (short)0, d00, false, false);
      d01 = __builtin_amdgcn_wmma_f32_16x16x32_bf16(false, fA0, false, fB1,
                                                    (short)0, d01, false, false);
      d10 = __builtin_amdgcn_wmma_f32_16x16x32_bf16(false, fA1, false, fB0,
                                                    (short)0, d10, false, false);
      d11 = __builtin_amdgcn_wmma_f32_16x16x32_bf16(false, fA1, false, fB1,
                                                    (short)0, d11, false, false);
    }
    float bias0 = Bhf[(br * DEPTH + l) * HID + n0];
    float bias1 = Bhf[(br * DEPTH + l) * HID + n1];
    if (l < DEPTH - 1) {
      #pragma unroll
      for (int v = 0; v < 8; ++v) {
        int M0 = v + hoff, M1 = 16 + v + hoff;
        lds_h[1 - bufin][M0 * HID + n0] = (__bf16)fmaxf(d00[v] + bias0, 0.f);
        lds_h[1 - bufin][M0 * HID + n1] = (__bf16)fmaxf(d01[v] + bias1, 0.f);
        lds_h[1 - bufin][M1 * HID + n0] = (__bf16)fmaxf(d10[v] + bias0, 0.f);
        lds_h[1 - bufin][M1 * HID + n1] = (__bf16)fmaxf(d11[v] + bias1, 0.f);
      }
      __syncthreads();
      bufin ^= 1;
    } else {
      // relu output >= 0 => leaky_relu is identity; scale by att and scatter
      #pragma unroll
      for (int v = 0; v < 8; ++v) {
        int M0 = v + hoff, M1 = 16 + v + hoff;
        float h00 = fmaxf(d00[v] + bias0, 0.f) * attb;
        float h01 = fmaxf(d01[v] + bias1, 0.f) * attb;
        float h10 = fmaxf(d10[v] + bias0, 0.f) * attb;
        float h11 = fmaxf(d11[v] + bias1, 0.f) * attb;
        if (s_i[M0] >= 0) {
          float* o = out + (size_t)s_i[M0] * HID;
          atomicAdd(o + n0, h00);
          atomicAdd(o + n1, h01);
        }
        if (s_i[M1] >= 0) {
          float* o = out + (size_t)s_i[M1] * HID;
          atomicAdd(o + n0, h10);
          atomicAdd(o + n1, h11);
        }
      }
    }
  }
}

// ---------------------------------------------------------------------------
extern "C" void kernel_launch(void* const* d_in, const int* in_sizes, int n_in,
                              void* d_out, int out_size, void* d_ws, size_t ws_size,
                              hipStream_t stream) {
  const float* node_feature = (const float*)d_in[0];
  const float* geo          = (const float*)d_in[1];
  const int*   eidx         = (const int*)  d_in[2];
  const int*   edx_ij       = (const int*)  d_in[3];
  const int*   edx_jk       = (const int*)  d_in[4];
  const float* att          = (const float*)d_in[5];
  const float* W0           = (const float*)d_in[6];
  const float* b0           = (const float*)d_in[7];
  const float* Wh           = (const float*)d_in[8];
  const float* bh           = (const float*)d_in[9];
  const float* gam          = (const float*)d_in[10];
  const float* bet          = (const float*)d_in[11];
  const float* rmu          = (const float*)d_in[12];
  const float* rvr          = (const float*)d_in[13];
  const int*   num_edge_in  = (const int*)  d_in[14];
  const int E = in_sizes[3];                  // edx_ij length

  __bf16* W0t = (__bf16*)d_ws;
  __bf16* Wht = W0t + W0T_ELEMS;
  float*  B0f = (float*)(Wht + WHT_ELEMS);
  float*  Bhf = B0f + B0F_ELEMS;
  int*    meta = (int*)(Bhf + BHF_ELEMS);     // 32 ints reserved
  int*    perm = meta + 32;                   // E ints

  // zero output accumulator + routing counters (capturable memset nodes)
  hipMemsetAsync(d_out, 0, (size_t)out_size * sizeof(float), stream);
  hipMemsetAsync(meta, 0, 8 * sizeof(int), stream);   // cnt[4], cursor[4]

  const int prepN = W0T_ELEMS + WHT_ELEMS + B0F_ELEMS + BHF_ELEMS;
  spnn_prep<<<(prepN + 255) / 256, 256, 0, stream>>>(
      W0, b0, Wh, bh, gam, bet, rmu, rvr, W0t, Wht, B0f, Bhf);

  const int eBlocks = (E + 255) / 256;
  route_count  <<<eBlocks, 256, 0, stream>>>(edx_ij, edx_jk, num_edge_in, meta, E);
  route_scan   <<<1, 1, 0, stream>>>(meta);
  route_scatter<<<eBlocks, 256, 0, stream>>>(edx_ij, edx_jk, num_edge_in,
                                             meta, perm, E);

  const int maxTiles = E / TILE_M + NBR;      // >= sum_br ceil(cnt[br]/TILE_M)
  spnn_fused<<<maxTiles, 256, 0, stream>>>(
      node_feature, geo, eidx, att, W0t, Wht, B0f, Bhf,
      meta, perm, (float*)d_out, E);
}
